// ModelTransformer_69466801045766
// MI455X (gfx1250) — matrix operand order repaired and checked by reading, not compiled, not run on hardware
//
#include <hip/hip_runtime.h>
#include <hip/hip_bf16.h>
#include <math.h>

typedef __attribute__((ext_vector_type(16))) __bf16 v16bf;
typedef __attribute__((ext_vector_type(8)))  float  v8f;

#define BKP 40  // padded LDS K-pitch in bf16 elems (80B = multiple of 16B)

// ---------------------------------------------------------------------------
// Generic batched GEMM: C = act(A * B + bias)
//  A: fp32 [M x K] row-major (lda)
//  B: fp32, bNK==0 -> memory is [K x N] row-major (weights / V)
//           bNK==1 -> memory is [N x K] row-major (K-matrix for Q*K^T)
//  Staged to LDS as bf16 (double-buffered, register-prefetched);
//  compute via v_wmma_f32_16x16x32_bf16, fp32 accumulate.
//  Per wave: 32(M) x 64(N) -> 8 WMMAs per 32-deep K step.
//  Batched over blockIdx.z: z -> (bb = z/H, hh = z%H) pointer strides.
// ---------------------------------------------------------------------------
template <int BM, int BN>
__global__ __launch_bounds__((BM / 32) * (BN / 64) * 32)
void gemm_bf16_wmma(const float* __restrict__ A, long lda, long aStrB, long aStrH,
                    const float* __restrict__ Bm, long ldb, long bStrB, long bStrH,
                    float* __restrict__ C, long ldc, long cStrB, long cStrH,
                    const float* __restrict__ bias,
                    int K, int H, int bNK, int relu)
{
    constexpr int WCOLS = BN / 64;
    constexpr int WROWS = BM / 32;
    constexpr int NT    = WROWS * WCOLS * 32;   // threads per block
    constexpr int A4    = (BM * 8) / NT;        // float4 loads per thread for A tile
    constexpr int B4    = (BN * 8) / NT;        // float4 loads per thread for B tile

    __shared__ __bf16 As[2][BM][BKP];
    __shared__ __bf16 Bs[2][BN][BKP];

    const int z  = blockIdx.z;
    const int bb = z / H, hh = z % H;
    A  += (long)bb * aStrB + (long)hh * aStrH;
    Bm += (long)bb * bStrB + (long)hh * bStrH;
    C  += (long)bb * cStrB + (long)hh * cStrH;

    const int m0   = blockIdx.y * BM;
    const int n0   = blockIdx.x * BN;
    const int tid  = threadIdx.x;
    const int lane = tid & 31;
    const int wave = tid >> 5;
    const int wr   = wave / WCOLS, wc = wave % WCOLS;
    const int lm   = lane & 15;
    const int half = lane >> 4;

    v8f acc[2][4] = {};          // [mi][ni] : 32x64 per wave
    float4 ra[A4], rb[B4];

    // ---- staging helpers (issue all loads, then convert+store) ----
    auto loadA = [&](int kt) {
#pragma unroll
        for (int j = 0; j < A4; ++j) {
            int idx = tid + j * NT;
            int r = idx >> 3, c = (idx & 7) << 2;
            ra[j] = *(const float4*)(A + (long)(m0 + r) * lda + (kt * 32 + c));
        }
    };
    auto storeA = [&](int buf) {
#pragma unroll
        for (int j = 0; j < A4; ++j) {
            int idx = tid + j * NT;
            int r = idx >> 3, c = (idx & 7) << 2;
            As[buf][r][c + 0] = (__bf16)ra[j].x; As[buf][r][c + 1] = (__bf16)ra[j].y;
            As[buf][r][c + 2] = (__bf16)ra[j].z; As[buf][r][c + 3] = (__bf16)ra[j].w;
        }
    };
    auto loadB = [&](int kt) {
        if (bNK) {
#pragma unroll
            for (int j = 0; j < B4; ++j) {
                int idx = tid + j * NT;
                int r = idx >> 3, c = (idx & 7) << 2;        // r = n, c = k
                rb[j] = *(const float4*)(Bm + (long)(n0 + r) * ldb + (kt * 32 + c));
            }
        } else {
#pragma unroll
            for (int j = 0; j < B4; ++j) {
                int idx = tid + j * NT;
                int r = idx / (BN / 4), c = (idx % (BN / 4)) << 2;   // r = k, c = n
                rb[j] = *(const float4*)(Bm + (long)(kt * 32 + r) * ldb + (n0 + c));
            }
        }
    };
    auto storeB = [&](int buf) {
        if (bNK) {
#pragma unroll
            for (int j = 0; j < B4; ++j) {
                int idx = tid + j * NT;
                int r = idx >> 3, c = (idx & 7) << 2;
                Bs[buf][r][c + 0] = (__bf16)rb[j].x; Bs[buf][r][c + 1] = (__bf16)rb[j].y;
                Bs[buf][r][c + 2] = (__bf16)rb[j].z; Bs[buf][r][c + 3] = (__bf16)rb[j].w;
            }
        } else {  // transpose into [n][k]
#pragma unroll
            for (int j = 0; j < B4; ++j) {
                int idx = tid + j * NT;
                int r = idx / (BN / 4), c = (idx % (BN / 4)) << 2;
                Bs[buf][c + 0][r] = (__bf16)rb[j].x;
                Bs[buf][c + 1][r] = (__bf16)rb[j].y;
                Bs[buf][c + 2][r] = (__bf16)rb[j].z;
                Bs[buf][c + 3][r] = (__bf16)rb[j].w;
            }
        }
    };

    const int nk = K >> 5;

    loadA(0); loadB(0);
    storeA(0); storeB(0);
    __syncthreads();

    for (int kt = 0; kt < nk; ++kt) {
        const int cur = kt & 1;
        if (kt + 1 < nk) { loadA(kt + 1); loadB(kt + 1); }   // prefetch (loads in flight)

        // ---- fragments per ISA 7.12.2 layouts (wave32) ----
        v16bf af[2], bfr[4];
#pragma unroll
        for (int mi = 0; mi < 2; ++mi) {
            int m  = wr * 32 + mi * 16 + lm;
            int c0 = half * 8;   // lanes<16: K0..7/K16..23 ; lanes>=16: K8..15/K24..31
#pragma unroll
            for (int j = 0; j < 8; ++j) {
                af[mi][j]     = As[cur][m][c0 + j];
                af[mi][8 + j] = As[cur][m][c0 + 16 + j];
            }
        }
#pragma unroll
        for (int ni = 0; ni < 4; ++ni) {
            int n  = wc * 64 + ni * 16 + lm;
            int ks = half * 8;   // lanes<16: K0..7/K16..23 ; lanes>=16: K8..15/K24..31
#pragma unroll
            for (int j = 0; j < 8; ++j) { bfr[ni][j] = Bs[cur][n][ks + j]; bfr[ni][8 + j] = Bs[cur][n][ks + 16 + j]; }
        }
#pragma unroll
        for (int mi = 0; mi < 2; ++mi)
#pragma unroll
            for (int ni = 0; ni < 4; ++ni)
                acc[mi][ni] = __builtin_amdgcn_wmma_f32_16x16x32_bf16(
                    false, af[mi], false, bfr[ni], (short)0, acc[mi][ni], false, false);

        if (kt + 1 < nk) {
            storeA(cur ^ 1); storeB(cur ^ 1);   // convert + store prefetched tile
            __syncthreads();
        }
    }

    // ---- epilogue: C/D layout (M = r + 8*(lane>=16), N = lane%16) ----
#pragma unroll
    for (int mi = 0; mi < 2; ++mi)
#pragma unroll
        for (int ni = 0; ni < 4; ++ni) {
            int gn = n0 + wc * 64 + ni * 16 + lm;
            float bv = bias ? bias[gn] : 0.0f;
#pragma unroll
            for (int r = 0; r < 8; ++r) {
                int gm = m0 + wr * 32 + mi * 16 + r + half * 8;
                float val = acc[mi][ni][r] + bv;
                if (relu) val = fmaxf(val, 0.0f);
                C[(long)gm * ldc + gn] = val;
            }
        }
}

// ---------------------------------------------------------------------------
// Row softmax with multiplicative mask (applied BEFORE softmax, as reference)
//  mode 0: none ; mode 1: causal(j<=i) & padding[b,i] ; mode 2: padding[b,i]
// ---------------------------------------------------------------------------
__global__ __launch_bounds__(256)
void softmax_mask_kernel(float* __restrict__ S, int Tq, int Tk, int H, int Tpad,
                         const unsigned char* __restrict__ pad, int mode, float scale)
{
    const int i = blockIdx.x;
    const int z = blockIdx.y;
    const int b = z / H;
    float* row = S + ((long)z * Tq + i) * Tk;
    __shared__ float sm[256];
    const int tid = threadIdx.x;

    float pq = 1.0f;
    if (mode != 0 && pad) pq = pad[(long)b * Tpad + i] ? 1.0f : 0.0f;

    float mx = -__builtin_inff();
    for (int j = tid; j < Tk; j += 256) {
        float msk = (mode == 1) ? ((j <= i) ? pq : 0.0f) : ((mode == 2) ? pq : 1.0f);
        mx = fmaxf(mx, row[j] * scale * msk);
    }
    sm[tid] = mx; __syncthreads();
    for (int s = 128; s > 0; s >>= 1) { if (tid < s) sm[tid] = fmaxf(sm[tid], sm[tid + s]); __syncthreads(); }
    mx = sm[0]; __syncthreads();

    float sum = 0.0f;
    for (int j = tid; j < Tk; j += 256) {
        float msk = (mode == 1) ? ((j <= i) ? pq : 0.0f) : ((mode == 2) ? pq : 1.0f);
        float e = expf(row[j] * scale * msk - mx);
        row[j] = e;
        sum += e;
    }
    sm[tid] = sum; __syncthreads();
    for (int s = 128; s > 0; s >>= 1) { if (tid < s) sm[tid] += sm[tid + s]; __syncthreads(); }
    float inv = 1.0f / sm[0];
    for (int j = tid; j < Tk; j += 256) row[j] *= inv;
}

// ---------------------------------------------------------------------------
// out = LayerNorm(Xa + Xb) * g + be   (biased var, eps = 1e-12). D == 1024.
// ---------------------------------------------------------------------------
__global__ __launch_bounds__(256)
void add_ln_kernel(const float* __restrict__ Xa, const float* __restrict__ Xb,
                   const float* __restrict__ g, const float* __restrict__ be,
                   float* __restrict__ out, int Dd)
{
    const long row = blockIdx.x;
    Xa += row * Dd; Xb += row * Dd; out += row * Dd;
    __shared__ float sm[256];
    const int tid = threadIdx.x;

    float x[4];
    float s1 = 0.f;
#pragma unroll
    for (int t = 0; t < 4; ++t) { int c = tid + t * 256; x[t] = Xa[c] + Xb[c]; s1 += x[t]; }
    sm[tid] = s1; __syncthreads();
    for (int s = 128; s > 0; s >>= 1) { if (tid < s) sm[tid] += sm[tid + s]; __syncthreads(); }
    float mean = sm[0] / (float)Dd; __syncthreads();

    float s2 = 0.f;
#pragma unroll
    for (int t = 0; t < 4; ++t) { float d = x[t] - mean; s2 += d * d; }
    sm[tid] = s2; __syncthreads();
    for (int s = 128; s > 0; s >>= 1) { if (tid < s) sm[tid] += sm[tid + s]; __syncthreads(); }
    float rstd = rsqrtf(sm[0] / (float)Dd + 1e-12f);
#pragma unroll
    for (int t = 0; t < 4; ++t) { int c = tid + t * 256; out[c] = g[c] * (x[t] - mean) * rstd + be[c]; }
}

// ---------------------------------------------------------------------------
// out = x + sinusoidal_pe(t, c)
// ---------------------------------------------------------------------------
__global__ __launch_bounds__(256)
void add_pe_kernel(const float* __restrict__ x, float* __restrict__ out,
                   int T_, int Dd, long total)
{
    long idx = (long)blockIdx.x * 256 + threadIdx.x;
    if (idx >= total) return;
    int c   = (int)(idx % Dd);
    long bt = idx / Dd;
    int t   = (int)(bt % T_);
    int pairb = c & ~1;
    float div = powf(10000.0f, (float)pairb / (float)Dd);
    float ang = (float)t / div;
    float pe  = (c & 1) ? cosf(ang) : sinf(ang);
    out[idx] = x[idx] + pe;
}

// ===========================================================================
// Host orchestration
// ===========================================================================
namespace {

enum { Bsz = 4, T = 1024, D = 1024, F = 4096, L = 4, NH = 16, DH = 64 };
const long BT = (long)Bsz * T;            // 4096
const long DD = (long)D * D;

struct AttnP { const float *wq, *bq, *wk, *bk, *wv, *bv, *wo, *bo; };

inline void gemm(hipStream_t s,
                 const float* A, long lda, long aB, long aH,
                 const float* Bm, long ldb, long bB, long bH,
                 float* C, long ldc, long cB, long cH,
                 const float* bias, int M, int N, int K,
                 int Hh, int Z, int bNK, int relu)
{
    if (N % 128 == 0) {
        dim3 g(N / 128, M / 128, Z), blk(256);
        hipLaunchKernelGGL((gemm_bf16_wmma<128, 128>), g, blk, 0, s,
                           A, lda, aB, aH, Bm, ldb, bB, bH, C, ldc, cB, cH,
                           bias, K, Hh, bNK, relu);
    } else {
        dim3 g(N / 64, M / 128, Z), blk(128);
        hipLaunchKernelGGL((gemm_bf16_wmma<128, 64>), g, blk, 0, s,
                           A, lda, aB, aH, Bm, ldb, bB, bH, C, ldc, cB, cH,
                           bias, K, Hh, bNK, relu);
    }
}

// plain (unbatched) projection: [BT x K] @ [K x N] + bias
inline void proj(hipStream_t s, const float* A, const float* W, const float* bias,
                 float* C, int N, int K, int relu)
{
    gemm(s, A, K, 0, 0, W, N, 0, 0, C, N, 0, 0, bias, (int)BT, N, K, 1, 1, 0, relu);
}

inline void ln(hipStream_t s, const float* a, const float* b,
               const float* g, const float* be, float* out)
{
    hipLaunchKernelGGL(add_ln_kernel, dim3((unsigned)BT), dim3(256), 0, s, a, b, g, be, out, D);
}

// Full multi-head attention: out = MHA(xq, xkv) projected by wo (+bo) -> t1
inline void run_mha(hipStream_t s, const float* xq, const float* xkv, const AttnP& p,
                    float* q, float* k, float* v, float* scores, float* t0, float* t1,
                    int mode, const unsigned char* pad)
{
    proj(s, xq,  p.wq, p.bq, q, D, D, 0);
    proj(s, xkv, p.wk, p.bk, k, D, D, 0);
    proj(s, xkv, p.wv, p.bv, v, D, D, 0);

    // S[b,h] = Q_slice @ K_slice^T   (K matrix is [T x DH] row-major = NK layout)
    gemm(s, q, D, (long)T * D, DH,
            k, D, (long)T * D, DH,
            scores, T, (long)NH * T * T, (long)T * T,
            nullptr, T, T, DH, NH, Bsz * NH, /*bNK=*/1, 0);

    hipLaunchKernelGGL(softmax_mask_kernel, dim3(T, Bsz * NH), dim3(256), 0, s,
                       scores, T, T, NH, T, pad, mode, 0.125f /*1/sqrt(64)*/);

    // O[b,h] = P @ V_slice  (V is [T x DH] with row stride D = KN layout)
    gemm(s, scores, T, (long)NH * T * T, (long)T * T,
            v, D, (long)T * D, DH,
            t0, D, (long)T * D, DH,
            nullptr, T, DH, T, NH, Bsz * NH, /*bNK=*/0, 0);

    proj(s, t0, p.wo, p.bo, t1, D, D, 0);
}

} // namespace

extern "C" void kernel_launch(void* const* d_in, const int* in_sizes, int n_in,
                              void* d_out, int out_size, void* d_ws, size_t ws_size,
                              hipStream_t stream)
{
    // ---- inputs (jax pytree order: dicts flattened by sorted key) ----
    const float* x_enc = (const float*)d_in[0];
    const float* x_dec = (const float*)d_in[1];
    const unsigned char* pad = (const unsigned char*)d_in[2];

    // enc_params, base 3, keys sorted:
    // b1 b2 bk bo bq bv f1b f1w f2b f2w g1 g2 wk wo wq wv
    const float* e_b1  = (const float*)d_in[3 + 0];
    const float* e_b2  = (const float*)d_in[3 + 1];
    const float* e_bk  = (const float*)d_in[3 + 2];
    const float* e_bo  = (const float*)d_in[3 + 3];
    const float* e_bq  = (const float*)d_in[3 + 4];
    const float* e_bv  = (const float*)d_in[3 + 5];
    const float* e_f1b = (const float*)d_in[3 + 6];
    const float* e_f1w = (const float*)d_in[3 + 7];
    const float* e_f2b = (const float*)d_in[3 + 8];
    const float* e_f2w = (const float*)d_in[3 + 9];
    const float* e_g1  = (const float*)d_in[3 + 10];
    const float* e_g2  = (const float*)d_in[3 + 11];
    const float* e_wk  = (const float*)d_in[3 + 12];
    const float* e_wo  = (const float*)d_in[3 + 13];
    const float* e_wq  = (const float*)d_in[3 + 14];
    const float* e_wv  = (const float*)d_in[3 + 15];

    // dec_params, base 19, keys sorted:
    // b1 b2 b3 ca_bk ca_bo ca_bq ca_bv ca_wk ca_wo ca_wq ca_wv
    // f1b f1w f2b f2w g1 g2 g3 sa_bk sa_bo sa_bq sa_bv sa_wk sa_wo sa_wq sa_wv
    const float* d_b1   = (const float*)d_in[19 + 0];
    const float* d_b2   = (const float*)d_in[19 + 1];
    const float* d_b3   = (const float*)d_in[19 + 2];
    const float* ca_bk  = (const float*)d_in[19 + 3];
    const float* ca_bo  = (const float*)d_in[19 + 4];
    const float* ca_bq  = (const float*)d_in[19 + 5];
    const float* ca_bv  = (const float*)d_in[19 + 6];
    const float* ca_wk  = (const float*)d_in[19 + 7];
    const float* ca_wo  = (const float*)d_in[19 + 8];
    const float* ca_wq  = (const float*)d_in[19 + 9];
    const float* ca_wv  = (const float*)d_in[19 + 10];
    const float* d_f1b  = (const float*)d_in[19 + 11];
    const float* d_f1w  = (const float*)d_in[19 + 12];
    const float* d_f2b  = (const float*)d_in[19 + 13];
    const float* d_f2w  = (const float*)d_in[19 + 14];
    const float* d_g1   = (const float*)d_in[19 + 15];
    const float* d_g2   = (const float*)d_in[19 + 16];
    const float* d_g3   = (const float*)d_in[19 + 17];
    const float* sa_bk  = (const float*)d_in[19 + 18];
    const float* sa_bo  = (const float*)d_in[19 + 19];
    const float* sa_bq  = (const float*)d_in[19 + 20];
    const float* sa_bv  = (const float*)d_in[19 + 21];
    const float* sa_wk  = (const float*)d_in[19 + 22];
    const float* sa_wo  = (const float*)d_in[19 + 23];
    const float* sa_wq  = (const float*)d_in[19 + 24];
    const float* sa_wv  = (const float*)d_in[19 + 25];

    // ---- workspace layout (fp32) ----
    float* ws = (float*)d_ws;
    float* xA     = ws;                       // [BT, D]
    float* xB     = xA + BT * D;
    float* q      = xB + BT * D;
    float* k      = q  + BT * D;
    float* v      = k  + BT * D;
    float* t0     = v  + BT * D;
    float* t1     = t0 + BT * D;
    float* yA     = t1 + BT * D;
    float* yB     = yA + BT * D;
    float* hbuf   = yB + BT * D;              // [BT, F]
    float* scores = hbuf + BT * (long)F;      // [B*NH, T, T]

    const long total = BT * (long)D;
    hipStream_t s = stream;

    // ================= Encoder =================
    hipLaunchKernelGGL(add_pe_kernel, dim3((unsigned)(total / 256)), dim3(256), 0, s,
                       x_enc, xA, T, D, total);

    for (int i = 0; i < L; ++i) {
        AttnP p = { e_wq + i * DD, e_bq + (long)i * D,
                    e_wk + i * DD, e_bk + (long)i * D,
                    e_wv + i * DD, e_bv + (long)i * D,
                    e_wo + i * DD, e_bo + (long)i * D };
        run_mha(s, xA, xA, p, q, k, v, scores, t0, t1, /*mode=*/0, nullptr);
        ln(s, xA, t1, e_g1 + (long)i * D, e_b1 + (long)i * D, xB);

        proj(s, xB, e_f1w + (long)i * D * F, e_f1b + (long)i * F, hbuf, F, D, /*relu=*/1);
        proj(s, hbuf, e_f2w + (long)i * F * D, e_f2b + (long)i * D, t1, D, F, 0);
        ln(s, xB, t1, e_g2 + (long)i * D, e_b2 + (long)i * D, xA);
    }
    float* mem = xA;   // encoder output

    // ================= Decoder (single layer, faithful to reference) =================
    hipLaunchKernelGGL(add_pe_kernel, dim3((unsigned)(total / 256)), dim3(256), 0, s,
                       x_dec, yA, T, D, total);

    AttnP psa = { sa_wq, sa_bq, sa_wk, sa_bk, sa_wv, sa_bv, sa_wo, sa_bo };
    run_mha(s, yA, yA, psa, q, k, v, scores, t0, t1, /*mode=*/1, pad);  // causal & padding
    ln(s, yA, t1, d_g1, d_b1, yB);

    AttnP pca = { ca_wq, ca_bq, ca_wk, ca_bk, ca_wv, ca_bv, ca_wo, ca_bo };
    run_mha(s, yB, mem, pca, q, k, v, scores, t0, t1, /*mode=*/2, pad); // padding broadcast
    ln(s, yB, t1, d_g2, d_b2, yA);

    proj(s, yA, d_f1w, d_f1b, hbuf, F, D, /*relu=*/1);
    proj(s, hbuf, d_f2w, d_f2b, t1, D, F, 0);
    ln(s, yA, t1, d_g3, d_b3, (float*)d_out);

    (void)in_sizes; (void)n_in; (void)out_size; (void)ws_size;
}
